// _StratsTransformer_84344567759589
// MI455X (gfx1250) — matrix operand order repaired and checked by reading, not compiled, not run on hardware
//
#include <hip/hip_runtime.h>
#include <hip/hip_bf16.h>
#include <math.h>

// ---------------------------------------------------------------------------
// Types for CDNA5 WMMA (wave32)
// ---------------------------------------------------------------------------
typedef __bf16 bf16_t;
typedef __attribute__((ext_vector_type(16))) __bf16 v16bf;
typedef __attribute__((ext_vector_type(8)))  __bf16 v8bf;
typedef __attribute__((ext_vector_type(8)))  float  v8f;
typedef __attribute__((ext_vector_type(4)))  unsigned int v4u;
typedef __attribute__((ext_vector_type(8)))  int v8i;
typedef __attribute__((ext_vector_type(4)))  int v4i;

// Problem constants (match reference)
#define NL   4
#define HH   8
#define DD   256
#define DKK  32
#define DFF  512
#define BB   4
#define LL   2048
#define MM   (BB * LL)   // 8192 rows

// GEMM epilogue flags
#define GF_BIAS       1
#define GF_GELU       2
#define GF_RESID_HALF 4
#define GF_OUT_F32    8
#define GF_OUT_BF16   16
#define GF_OUT_BF16T  32

static __device__ __forceinline__ v8f wmma_bf16(v16bf a, v16bf b, v8f c) {
  // D = A(16x32 bf16) * B(32x16 bf16) + C(16x16 f32)
  return __builtin_amdgcn_wmma_f32_16x16x32_bf16(
      /*neg_a=*/false, a, /*neg_b=*/false, b,
      /*c_mod=*/(short)0, c, /*reuse_a=*/false, /*reuse_b=*/false);
}

// A-fragment (16x32, bf16, row-major source, ld in elements).
// ISA layout: lane t (t<16 -> K-half 0, t>=16 -> K-half 1), row = t&15.
static __device__ __forceinline__ v16bf load_a_frag(const bf16_t* base, int ld) {
  const int lane = threadIdx.x & 31;
  const int row  = lane & 15;
  const int koff = (lane >> 4) << 3;   // 0 or 8
  const bf16_t* p = base + (size_t)row * ld + koff;
  v8bf lo = *(const v8bf*)(p);         // K = koff .. koff+7
  v8bf hi = *(const v8bf*)(p + 16);    // K = 16+koff .. 16+koff+7
  v16bf r;
#pragma unroll
  for (int i = 0; i < 8; ++i) { r[i] = lo[i]; r[8 + i] = hi[i]; }
  return r;
}

// B-fragment (32x16, bf16). Source stored N-major: src[n*ld + k].
// ISA layout: lanes 0-15 hold K=0..7,16..23 of col n=lane, lanes 16-31 hold K=8..15,24..31.
static __device__ __forceinline__ v16bf load_b_frag(const bf16_t* base, size_t ld) {
  const int lane = threadIdx.x & 31;
  const int col  = lane & 15;
  const int koff = (lane >> 4) << 3;   // 0 or 8
  const bf16_t* p = base + (size_t)col * ld + koff;
  v8bf lo = *(const v8bf*)(p);
  v8bf hi = *(const v8bf*)(p + 16);
  v16bf r;
#pragma unroll
  for (int i = 0; i < 8; ++i) { r[i] = lo[i]; r[8 + i] = hi[i]; }
  return r;
}

// ---------------------------------------------------------------------------
// Tensor Data Mover: DMA a [rows x K] bf16 panel (row stride = K elems) from
// global memory into LDS.  Builds a D# per CDNA5 ISA ch.8 (2-D tile,
// data_size=2B, count=1, type=2) and issues TENSOR_LOAD_TO_LDS (wave-level,
// EXEC ignored).  Caller must s_wait_tensorcnt + barrier before consuming.
// This toolchain's builtin takes 6 args (g0 4xu32, g1 8xi32, g2 4xi32,
// g3 4xi32, extra 8xi32, cpol) -- groups 2/3 unused for a 2-D tensor.
// ---------------------------------------------------------------------------
static __device__ __forceinline__ void tdm_load_panel_to_lds(
    const bf16_t* gsrc, bf16_t* ldst, int K, int rows) {
  const unsigned long long ga = (unsigned long long)(uintptr_t)gsrc;
  const unsigned lds_off = (unsigned)(uintptr_t)ldst;  // low 32 bits = LDS offset

  v4u g0;
  g0[0] = 1u;                                   // count=1, is_restore=0, gather=0
  g0[1] = lds_off;                              // lds_addr
  g0[2] = (unsigned)(ga & 0xffffffffu);         // global_addr[31:0]
  g0[3] = (unsigned)((ga >> 32) & 0x01ffffffu)  // global_addr[56:32]
          | 0x80000000u;                        // type=2 ("image") in [127:126]

  v8i g1;
  g1[0] = 0x00010000;                           // workgroup_mask=0, data_size=1 (2B)
  g1[1] = (int)((unsigned)(K & 0xffff) << 16);  // tensor_dim0[15:0] (bits 63:48)
  g1[2] = (int)(((unsigned)K >> 16) | ((unsigned)(rows & 0xffff) << 16)); // dim0 hi | dim1 lo
  g1[3] = (int)(((unsigned)rows >> 16) | ((unsigned)(K & 0xffff) << 16)); // dim1 hi | tile_dim0
  g1[4] = (int)((unsigned)(rows & 0xffff));     // tile_dim1 | tile_dim2=0
  g1[5] = (int)(unsigned)K;                     // tensor_dim0_stride[31:0]
  g1[6] = 0;                                    // stride0 hi | stride1 lo (unused, 2-D)
  g1[7] = 0;                                    // stride1 hi

  v4i gz4 = (v4i){0, 0, 0, 0};                  // groups 2/3 unused (<=2-D tensor)
  v8i gz8 = (v8i){0, 0, 0, 0, 0, 0, 0, 0};
  __builtin_amdgcn_tensor_load_to_lds(g0, g1, gz4, gz4, gz8, 0);
}

// ---------------------------------------------------------------------------
// Weight prep: f32 [rows,cols] (batched) -> bf16 transposed [cols,rows]
// ---------------------------------------------------------------------------
__global__ void k_transpose_bf16(const float* __restrict__ src, bf16_t* __restrict__ dst,
                                 int rows, int cols, long sstride, long dstride) {
  const int bat = blockIdx.z;
  const int idx = blockIdx.x * blockDim.x + threadIdx.x;
  const int total = rows * cols;
  if (idx >= total) return;
  const int r = idx / cols, c = idx % cols;
  dst[(size_t)bat * dstride + (size_t)c * rows + r] =
      (bf16_t)src[(size_t)bat * sstride + (size_t)r * cols + c];
}

__global__ void k_cvt_bf16(const float* __restrict__ s, bf16_t* __restrict__ d, int n) {
  const int i = blockIdx.x * blockDim.x + threadIdx.x;
  if (i < n) d[i] = (bf16_t)s[i];
}

// ---------------------------------------------------------------------------
// bf16 WMMA GEMM:  Y[M,N] = A[M,K] * W^T  (W stored [N,K] bf16)
// Block: 256 threads = 8 waves. Block tile 256(M) x 64(N); each wave computes
// a 32x64 strip (2 A-fragments x 4 B-fragments = 8 WMMAs per K-step).
// The B panel ([64,K] bf16, <=64KB) is DMA'd to LDS once per block by the
// Tensor Data Mover; the K-loop reads B with ds_load_b128 while A streams
// from global with double buffering.
// ---------------------------------------------------------------------------
__global__ __launch_bounds__(256) void k_gemm_wmma(
    const bf16_t* __restrict__ A,    // [M,K] row-major bf16
    const bf16_t* __restrict__ Bw,   // [N,K] row-major bf16 (transposed weight)
    const float*  __restrict__ bias, // [N] (GF_BIAS)
    const float*  __restrict__ resid,// [M,N] (GF_RESID_HALF)
    float* __restrict__ outF,        // [M,N] (GF_OUT_F32)
    bf16_t* __restrict__ outBF,      // [M,N] or [N,M] (GF_OUT_BF16 / _BF16T)
    int M, int N, int K, int flags)
{
  extern __shared__ bf16_t ldsB[];   // [64][K]

  const int wave = threadIdx.x >> 5;
  const int lane = threadIdx.x & 31;
  const int m0 = blockIdx.y * 256 + wave * 32;
  const int n0 = blockIdx.x * 64;

  // ---- TDM: stage this block's 64-column weight panel into LDS ----
  if (wave == 0) {
    tdm_load_panel_to_lds(Bw + (size_t)n0 * K, ldsB, K, 64);
    __builtin_amdgcn_s_wait_tensorcnt(0);
  }
  __syncthreads();

  v8f acc[2][4];
#pragma unroll
  for (int t = 0; t < 2; ++t)
#pragma unroll
    for (int j = 0; j < 4; ++j)
      acc[t][j] = (v8f){0.f,0.f,0.f,0.f,0.f,0.f,0.f,0.f};

  const bf16_t* aptr = A + (size_t)m0 * K;
  v16bf a0 = load_a_frag(aptr, K);
  v16bf a1 = load_a_frag(aptr + (size_t)16 * K, K);

  for (int k = 0; k < K; k += 32) {
    v16bf na0, na1;
    if (k + 32 < K) {                       // issue next A tiles early
      __builtin_prefetch((const void*)(aptr + k + 64), 0, 3);
      na0 = load_a_frag(aptr + k + 32, K);
      na1 = load_a_frag(aptr + (size_t)16 * K + k + 32, K);
    }
#pragma unroll
    for (int j = 0; j < 4; ++j) {
      v16bf bf = load_b_frag(ldsB + (size_t)(16 * j) * K + k, (size_t)K);
      acc[0][j] = wmma_bf16(a0, bf, acc[0][j]);
      acc[1][j] = wmma_bf16(a1, bf, acc[1][j]);
    }
    a0 = na0; a1 = na1;
  }

  // Epilogue. C/D layout: VGPR v holds row base + v + 8*(lane>=16), col = lane&15.
  const int rbase = (lane >> 4) << 3;
  const int col   = lane & 15;
#pragma unroll
  for (int t = 0; t < 2; ++t) {
#pragma unroll
    for (int j = 0; j < 4; ++j) {
      const int n = n0 + 16 * j + col;
      const float bv = (flags & GF_BIAS) ? bias[n] : 0.0f;
#pragma unroll
      for (int v = 0; v < 8; ++v) {
        const int m = m0 + 16 * t + rbase + v;
        float x = acc[t][j][v] + bv;
        if (flags & GF_GELU)       x = 0.5f * x * (1.0f + erff(x * 0.70710678118654752f));
        if (flags & GF_RESID_HALF) x = (x + resid[(size_t)m * N + n]) * 0.5f;
        if (flags & GF_OUT_F32)    outF[(size_t)m * N + n] = x;
        if (flags & GF_OUT_BF16)   outBF[(size_t)m * N + n] = (bf16_t)x;
        if (flags & GF_OUT_BF16T)  outBF[(size_t)n * M + m] = (bf16_t)x;
      }
    }
  }
}

// ---------------------------------------------------------------------------
// Flash attention (no 1/sqrt(dk) scale, padding mask, per reference).
// One wave per (b, h, 16-query tile). S tile = Q(16x32) x K^T(32x16) -> one
// WMMA per 16 keys; online softmax via LDS; P re-laid out through LDS into an
// A-fragment; O accumulated with 2 WMMAs per 32-key chunk.
// ---------------------------------------------------------------------------
__global__ __launch_bounds__(32) void k_flash_attn(
    const bf16_t* __restrict__ Q,   // [B*L, D] bf16 row-major
    const bf16_t* __restrict__ Km,  // [B*L, D] bf16 row-major
    const bf16_t* __restrict__ Vt,  // [D, B*L] bf16 (feature-major)
    const float*  __restrict__ mask,// [B, L]
    bf16_t* __restrict__ O)         // [B*L, D] bf16 row-major
{
  __shared__ float  lds_S[16 * 32];
  __shared__ bf16_t lds_P[16 * 32];
  __shared__ float  lds_alpha[16];
  __shared__ float  lds_scale[16];
  __shared__ float  lds_mk[32];

  const int lane = threadIdx.x;
  const int q0 = blockIdx.x * 16;
  const int h  = blockIdx.y;
  const int b  = blockIdx.z;
  const size_t rowBase = (size_t)b * LL + q0;

  const v16bf aq = load_a_frag(Q + rowBase * DD + h * DKK, DD);

  v8f o0 = (v8f){0.f,0.f,0.f,0.f,0.f,0.f,0.f,0.f};
  v8f o1 = o0;
  const v8f zero = o0;
  const int rbase = (lane >> 4) << 3;   // accumulator row base (0 or 8)
  const int ccol  = lane & 15;

  float mrun = -3.0e38f, lrun = 0.0f, mq = 0.0f;
  if (lane < 16) mq = mask[(size_t)b * LL + q0 + lane];

  for (int k0 = 0; k0 < LL; k0 += 32) {
    // key-side padding mask for this chunk
    lds_mk[lane] = mask[(size_t)b * LL + k0 + lane];

    // ---- S = Q * K^T (two 16x16 tiles) ----
    const bf16_t* kbase = Km + ((size_t)b * LL + k0) * DD + h * DKK;
    v16bf bk0 = load_b_frag(kbase, (size_t)DD);
    v16bf bk1 = load_b_frag(kbase + (size_t)16 * DD, (size_t)DD);
    v8f s0 = wmma_bf16(aq, bk0, zero);
    v8f s1 = wmma_bf16(aq, bk1, zero);

    // spill S tile to LDS in row-major [16][32]
#pragma unroll
    for (int v = 0; v < 8; ++v) {
      lds_S[(rbase + v) * 32 + ccol]      = s0[v];
      lds_S[(rbase + v) * 32 + 16 + ccol] = s1[v];
    }
    __syncthreads();

    // ---- online softmax: lanes 0..15 own one query row each ----
    if (lane < 16) {
      float mc = -3.0e38f;
#pragma unroll
      for (int j = 0; j < 32; ++j) {
        float s = lds_S[lane * 32 + j];
        s = (mq * lds_mk[j] > 0.0f) ? s : -3.0e38f;
        mc = fmaxf(mc, s);
      }
      const float mnew  = fmaxf(mrun, mc);
      const float alpha = __expf(mrun - mnew);
      float psum = 0.0f;
#pragma unroll
      for (int j = 0; j < 32; ++j) {
        float s = lds_S[lane * 32 + j];
        s = (mq * lds_mk[j] > 0.0f) ? s : -3.0e38f;
        const float p = __expf(s - mnew);
        psum += p;
        lds_P[lane * 32 + j] = (bf16_t)p;
      }
      lrun = lrun * alpha + psum;
      mrun = mnew;
      lds_alpha[lane] = alpha;
    }
    __syncthreads();

    // ---- O = diag(alpha) * O + P * V ----
    v16bf ap = load_a_frag(lds_P, 32);
    const bf16_t* vbase = Vt + (size_t)(h * DKK) * MM + (size_t)b * LL + k0;
    v16bf bv0 = load_b_frag(vbase, (size_t)MM);                    // features 0..15
    v16bf bv1 = load_b_frag(vbase + (size_t)16 * MM, (size_t)MM);  // features 16..31
#pragma unroll
    for (int v = 0; v < 8; ++v) {
      const float a = lds_alpha[rbase + v];
      o0[v] *= a; o1[v] *= a;
    }
    o0 = wmma_bf16(ap, bv0, o0);
    o1 = wmma_bf16(ap, bv1, o1);
    __syncthreads();
  }

  if (lane < 16) lds_scale[lane] = 1.0f / lrun;
  __syncthreads();
#pragma unroll
  for (int v = 0; v < 8; ++v) {
    const float sc = lds_scale[rbase + v];
    const size_t m = rowBase + rbase + v;
    O[m * DD + h * DKK + ccol]      = (bf16_t)(o0[v] * sc);
    O[m * DD + h * DKK + 16 + ccol] = (bf16_t)(o1[v] * sc);
  }
}

// ---------------------------------------------------------------------------
// Host-side orchestration
// ---------------------------------------------------------------------------
extern "C" void kernel_launch(void* const* d_in, const int* in_sizes, int n_in,
                              void* d_out, int out_size, void* d_ws, size_t ws_size,
                              hipStream_t stream) {
  const float* x    = (const float*)d_in[0];
  const float* mask = (const float*)d_in[1];
  const float* Wq   = (const float*)d_in[2];
  const float* Wk   = (const float*)d_in[3];
  const float* Wv   = (const float*)d_in[4];
  const float* Wo   = (const float*)d_in[5];
  const float* W1   = (const float*)d_in[6];
  const float* b1   = (const float*)d_in[7];
  const float* W2   = (const float*)d_in[8];
  const float* W2b  = (const float*)d_in[9];

  // ---- workspace carve-up (bytes) ----
  char* ws = (char*)d_ws;
  size_t off = 0;
  auto carve = [&](size_t bytes) { char* p = ws + off; off += (bytes + 255) & ~(size_t)255; return p; };

  const size_t perLayerWT = 3 * 256 * 256 + 256 * 256 + 256 * 512 + 512 * 256; // elems
  bf16_t* wT   = (bf16_t*)carve(NL * perLayerWT * sizeof(bf16_t));
  bf16_t* Xbf  = (bf16_t*)carve((size_t)MM * DD  * sizeof(bf16_t));
  bf16_t* Qbf  = (bf16_t*)carve((size_t)MM * DD  * sizeof(bf16_t));
  bf16_t* Kbf  = (bf16_t*)carve((size_t)MM * DD  * sizeof(bf16_t));
  bf16_t* Vtb  = (bf16_t*)carve((size_t)DD * MM  * sizeof(bf16_t));
  bf16_t* Obf  = (bf16_t*)carve((size_t)MM * DD  * sizeof(bf16_t));
  bf16_t* X1bf = (bf16_t*)carve((size_t)MM * DD  * sizeof(bf16_t));
  bf16_t* FFbf = (bf16_t*)carve((size_t)MM * DFF * sizeof(bf16_t));
  float*  xcA  = (float*)carve((size_t)MM * DD * sizeof(float));
  float*  xcB  = (float*)carve((size_t)MM * DD * sizeof(float));
  (void)ws_size; (void)in_sizes; (void)n_in; (void)out_size;

  // per-layer transposed-weight bases (elems within wT)
  auto wqT = [&](int n) { return wT + (size_t)n * perLayerWT; };
  auto wkT = [&](int n) { return wqT(n) + 256 * 256; };
  auto wvT = [&](int n) { return wkT(n) + 256 * 256; };
  auto woT = [&](int n) { return wvT(n) + 256 * 256; };
  auto w1T = [&](int n) { return woT(n) + 256 * 256; };
  auto w2T = [&](int n) { return w1T(n) + 256 * 512; };

  // ---- weight prep: f32 -> bf16 transposed [N_out, K_in] ----
  for (int n = 0; n < NL; ++n) {
    // Wq/Wk/Wv: [H, D, DK] -> per head transpose [DK, D]; out col = h*DK+e
    dim3 gq((256 * 32 + 255) / 256, 1, HH);
    k_transpose_bf16<<<gq, 256, 0, stream>>>(Wq + (size_t)n * HH * DD * DKK, wqT(n),
                                             DD, DKK, (long)DD * DKK, (long)DKK * DD);
    k_transpose_bf16<<<gq, 256, 0, stream>>>(Wk + (size_t)n * HH * DD * DKK, wkT(n),
                                             DD, DKK, (long)DD * DKK, (long)DKK * DD);
    k_transpose_bf16<<<gq, 256, 0, stream>>>(Wv + (size_t)n * HH * DD * DKK, wvT(n),
                                             DD, DKK, (long)DD * DKK, (long)DKK * DD);
    dim3 go((256 * 256 + 255) / 256, 1, 1);
    k_transpose_bf16<<<go, 256, 0, stream>>>(Wo + (size_t)n * DD * DD, woT(n),
                                             DD, DD, 0, 0);
    dim3 g1((256 * 512 + 255) / 256, 1, 1);
    k_transpose_bf16<<<g1, 256, 0, stream>>>(W1 + (size_t)n * DD * DFF, w1T(n),
                                             DD, DFF, 0, 0);
    k_transpose_bf16<<<g1, 256, 0, stream>>>(W2 + (size_t)n * DFF * DD, w2T(n),
                                             DFF, DD, 0, 0);
  }

  // initial activation -> bf16
  k_cvt_bf16<<<(MM * DD + 255) / 256, 256, 0, stream>>>(x, Xbf, MM * DD);

  const dim3 blk(256);
  const dim3 gN256(256 / 64, MM / 256);  // N=256 panels, 256-row block tiles
  const dim3 gN512(512 / 64, MM / 256);  // N=512
  const dim3 gFlash(LL / 16, HH, BB);
  const size_t lds256 = (size_t)64 * 256 * sizeof(bf16_t);  // 32 KB
  const size_t lds512 = (size_t)64 * 512 * sizeof(bf16_t);  // 64 KB

  for (int n = 0; n < NL; ++n) {
    const bool last = (n == NL - 1);
    const float* xcur = (n == 0) ? x : xcA;   // f32 residual input for this layer

    // Q, K (bf16 row-major), V (bf16 transposed)
    k_gemm_wmma<<<gN256, blk, lds256, stream>>>(Xbf, wqT(n), nullptr, nullptr, nullptr, Qbf,
                                                MM, 256, 256, GF_OUT_BF16);
    k_gemm_wmma<<<gN256, blk, lds256, stream>>>(Xbf, wkT(n), nullptr, nullptr, nullptr, Kbf,
                                                MM, 256, 256, GF_OUT_BF16);
    k_gemm_wmma<<<gN256, blk, lds256, stream>>>(Xbf, wvT(n), nullptr, nullptr, nullptr, Vtb,
                                                MM, 256, 256, GF_OUT_BF16T);

    k_flash_attn<<<gFlash, 32, 0, stream>>>(Qbf, Kbf, Vtb, mask, Obf);

    // xc1 = (O @ Wo + xcur) * 0.5  (f32 in xcB, bf16 in X1bf)
    k_gemm_wmma<<<gN256, blk, lds256, stream>>>(Obf, woT(n), nullptr, xcur, xcB, X1bf,
                                                MM, 256, 256,
                                                GF_RESID_HALF | GF_OUT_F32 | GF_OUT_BF16);

    // ff = gelu(xc1 @ W1 + b1)  (bf16)
    k_gemm_wmma<<<gN512, blk, lds256, stream>>>(X1bf, w1T(n), b1 + (size_t)n * DFF,
                                                nullptr, nullptr, FFbf,
                                                MM, 512, 256, GF_BIAS | GF_GELU | GF_OUT_BF16);

    // out = (ff @ W2 + b2 + xc1) * 0.5 -> f32 (d_out on last layer), bf16 for next
    float* outF = last ? (float*)d_out : xcA;
    int    fl   = GF_BIAS | GF_RESID_HALF | GF_OUT_F32 | (last ? 0 : GF_OUT_BF16);
    k_gemm_wmma<<<gN256, blk, lds512, stream>>>(FFbf, w2T(n), W2b + (size_t)n * DD,
                                                xcB, outF, last ? nullptr : Xbf,
                                                MM, 256, 512, fl);
  }
}
